// LinearAttention_16664473108965
// MI455X (gfx1250) — matrix operand order repaired and checked by reading, not compiled, not run on hardware
//
#include <hip/hip_runtime.h>
#include <stdint.h>

// ---------- types for WMMA ----------
typedef __attribute__((ext_vector_type(16))) __bf16 v16bf;
typedef __attribute__((ext_vector_type(8)))  float  v8f;

struct U4x2 { uint4 lo, hi; };   // 32 bytes == one v16bf fragment

__device__ __forceinline__ unsigned short f2bf(float f) {
  union { float f; uint32_t u; } v; v.f = f;
  uint32_t r = v.u + 0x7FFFu + ((v.u >> 16) & 1u);   // round-to-nearest-even
  return (unsigned short)(r >> 16);
}
__device__ __forceinline__ float bf2f(unsigned short h) {
  union { uint32_t u; float f; } v; v.u = ((uint32_t)h) << 16;
  return v.f;
}
// low 32 bits of a generic LDS pointer == LDS byte offset (ISA 10.2 aperture map)
__device__ __forceinline__ uint32_t lds_addr32(const void* p) {
  return (uint32_t)(uintptr_t)p;
}

// ---------- problem constants ----------
#define Bc   4
#define Sc   4096
#define HIDc 2048
#define Hc   16
#define DHc  128
#define DIMc 64
#define ROWS (Bc * Sc)          // 16384
#define KDq  (Hc * DIMc)        // 1024  (feature dim of Qf/Kf)

// ============================================================
// elementwise cast f32 -> bf16
// ============================================================
__global__ void cast_f32_bf16(const float* __restrict__ src,
                              unsigned short* __restrict__ dst, int n) {
  int i = blockIdx.x * blockDim.x + threadIdx.x;
  if (i < n) dst[i] = f2bf(src[i]);
}

__global__ void zero_f32(float* __restrict__ p, int n) {
  int i = blockIdx.x * blockDim.x + threadIdx.x;
  if (i < n) p[i] = 0.f;
}

// ============================================================
// fold feature-map into projection weight:
// eff[(h*DIM+m)*HID + k] = sum_d wmap[m*DH+d] * w[(h*DH+d)*HID + k]   (bf16 out)
// ============================================================
__global__ void fold_weight(const float* __restrict__ w,
                            const float* __restrict__ wmap,
                            unsigned short* __restrict__ eff) {
  int idx = blockIdx.x * blockDim.x + threadIdx.x;   // KDq * HIDc threads
  int k  = idx & (HIDc - 1);
  int o2 = idx >> 11;                                // 0..1023
  int h = o2 >> 6, m = o2 & 63;
  const float* wr = w + (size_t)(h * DHc) * HIDc + k;
  const float* mp = wmap + m * DHc;
  float acc = 0.f;
#pragma unroll 8
  for (int d = 0; d < DHc; ++d) acc += mp[d] * wr[(size_t)d * HIDc];
  eff[idx] = f2bf(acc);
}

__global__ void fold_bias(const float* __restrict__ b,
                          const float* __restrict__ wmap,
                          float* __restrict__ beff) {
  int idx = blockIdx.x * blockDim.x + threadIdx.x;   // 1024 threads
  if (idx >= KDq) return;
  int h = idx >> 6, m = idx & 63;
  float acc = 0.f;
#pragma unroll 8
  for (int d = 0; d < DHc; ++d) acc += wmap[m * DHc + d] * b[h * DHc + d];
  beff[idx] = acc;
}

// ============================================================
// NT bf16 WMMA GEMM:  C[M,N] = A[M,K] * W[N,K]^T   (+ epilogue)
// 2x4 register blocking: each wave owns a 32x64 C strip (8 WMMA tiles),
// 8 waves / block.  tilesN = N/64.
// EPI 0: y = elu(x+bias)+1 -> bf16    EPI 1: y = x+bias -> bf16
// EPI 2: y = x+bias -> f32
// ============================================================
template <int EPI>
__global__ __launch_bounds__(256) void gemm_nt(
    const unsigned short* __restrict__ A, int lda,
    const unsigned short* __restrict__ W, int ldb,
    const float* __restrict__ bias,
    void* __restrict__ Cout, int ldc,
    int tilesN, int Kdim) {
  const int lane = threadIdx.x & 31;
  const int widx = blockIdx.x * 8 + (threadIdx.x >> 5);
  const int tm = widx / tilesN, tn = widx % tilesN;
  const int l15 = lane & 15;
  const int hi  = (lane >> 4) & 1;

  // A fragment rows: tm*32 + {0,16} + l15 ; two 16B chunks at k0+8*hi, k0+16+8*hi
  const unsigned short* ap0 = A + (size_t)(tm * 32 + l15) * lda + hi * 8;
  const unsigned short* ap1 = ap0 + (size_t)16 * lda;
  // B fragment rows (= output cols): tn*64 + j*16 + l15 ; two 16B chunks at k0+8*hi, k0+16+8*hi
  const unsigned short* bp0 = W + (size_t)(tn * 64 + l15) * ldb + hi * 8;

  v8f acc[2][4] = {};
  for (int k0 = 0; k0 < Kdim; k0 += 32) {
    __builtin_prefetch(ap0 + k0 + 512, 0, 1);
    __builtin_prefetch(bp0 + k0 + 512, 0, 1);
    v16bf af[2], bfr[4];
    {
      U4x2 u;
      u.lo = *(const uint4*)(ap0 + k0);
      u.hi = *(const uint4*)(ap0 + k0 + 16);
      af[0] = __builtin_bit_cast(v16bf, u);
      u.lo = *(const uint4*)(ap1 + k0);
      u.hi = *(const uint4*)(ap1 + k0 + 16);
      af[1] = __builtin_bit_cast(v16bf, u);
    }
#pragma unroll
    for (int j = 0; j < 4; ++j) {
      const unsigned short* bp = bp0 + (size_t)(j * 16) * ldb;
      U4x2 u;
      u.lo = *(const uint4*)(bp + k0);
      u.hi = *(const uint4*)(bp + k0 + 16);
      bfr[j] = __builtin_bit_cast(v16bf, u);
    }
#pragma unroll
    for (int i = 0; i < 2; ++i)
#pragma unroll
      for (int j = 0; j < 4; ++j)
        acc[i][j] = __builtin_amdgcn_wmma_f32_16x16x32_bf16(
            false, af[i], false, bfr[j], (short)0, acc[i][j], false, false);
  }

#pragma unroll
  for (int j = 0; j < 4; ++j) {
    const int n = tn * 64 + j * 16 + l15;
    const float bval = bias[n];
#pragma unroll
    for (int i = 0; i < 2; ++i) {
      const int row0 = tm * 32 + i * 16 + hi * 8;
      if (EPI == 0) {
        unsigned short* C = (unsigned short*)Cout;
#pragma unroll
        for (int v = 0; v < 8; ++v) {
          float x = acc[i][j][v] + bval;
          float y = (x > 0.f) ? (x + 1.f) : __expf(x);   // elu(x)+1
          C[(size_t)(row0 + v) * ldc + n] = f2bf(y);
        }
      } else if (EPI == 1) {
        unsigned short* C = (unsigned short*)Cout;
#pragma unroll
        for (int v = 0; v < 8; ++v)
          C[(size_t)(row0 + v) * ldc + n] = f2bf(acc[i][j][v] + bval);
      } else {
        float* C = (float*)Cout;
#pragma unroll
        for (int v = 0; v < 8; ++v)
          C[(size_t)(row0 + v) * ldc + n] = acc[i][j][v] + bval;
      }
    }
  }
}

// ============================================================
// kv^T + ksum:  kvT[b,h,d,m] += sum_s V[s, h*DH+d] * Kf[s, h*DIM+m]
//               ksum[b,h,m]  += sum_s Kf[s, h*DIM+m]
// Staging via gfx1250 async global->LDS copies (ASYNCcnt tracked),
// f32 global atomics for the cross-block partials.
// ============================================================
__global__ __launch_bounds__(256) void kv_ksum(
    const unsigned short* __restrict__ Kf,    // [ROWS, KDq]
    const unsigned short* __restrict__ Vbf,   // [ROWS, HIDc]
    float* __restrict__ kvT,                  // [B,H,DH,DIM]
    float* __restrict__ ksum) {               // [B,H,DIM]
  __shared__ unsigned short Ks[128][DIMc];    // 16 KB
  __shared__ unsigned short Vs[128][DHc];     // 32 KB
  const int bh = blockIdx.x >> 5;             // (b*H + h), 64 total
  const int sb = blockIdx.x & 31;             // s-block within batch
  const int b  = bh >> 4, h = bh & 15;
  const int s0 = b * Sc + sb * 128;
  const int t  = threadIdx.x;

  // async stage K block: 128 rows x 8 uint4
  for (int i = t; i < 128 * 8; i += 256) {
    int r = i >> 3, c = i & 7;
    uint32_t lds = lds_addr32(&Ks[r][0]) + c * 16;
    const unsigned short* g = Kf + (size_t)(s0 + r) * KDq + h * DIMc + c * 8;
    asm volatile("global_load_async_to_lds_b128 %0, %1, off"
                 :: "v"(lds), "v"(g) : "memory");
  }
  // async stage V block: 128 rows x 16 uint4
  for (int i = t; i < 128 * 16; i += 256) {
    int r = i >> 4, c = i & 15;
    uint32_t lds = lds_addr32(&Vs[r][0]) + c * 16;
    const unsigned short* g = Vbf + (size_t)(s0 + r) * HIDc + h * DHc + c * 8;
    asm volatile("global_load_async_to_lds_b128 %0, %1, off"
                 :: "v"(lds), "v"(g) : "memory");
  }
  asm volatile("s_wait_asynccnt 0" ::: "memory");
  __syncthreads();

  const int d  = t & 127;
  const int mg = t >> 7;                      // 0..1 -> m in [mg*32, mg*32+32)
  float acc[32];
#pragma unroll
  for (int j = 0; j < 32; ++j) acc[j] = 0.f;

  for (int s = 0; s < 128; ++s) {
    float vv = bf2f(Vs[s][d]);
    const uint32_t* kp = (const uint32_t*)&Ks[s][mg * 32];
#pragma unroll
    for (int j2 = 0; j2 < 16; ++j2) {
      uint32_t pk = kp[j2];
      acc[2 * j2]     += vv * bf2f((unsigned short)(pk & 0xFFFFu));
      acc[2 * j2 + 1] += vv * bf2f((unsigned short)(pk >> 16));
    }
  }
  float* dst = kvT + (((size_t)bh * DHc + d) * DIMc + mg * 32);
#pragma unroll
  for (int j = 0; j < 32; ++j) atomicAdd(dst + j, acc[j]);

  if (t < DIMc) {                             // ksum partial for this s-block
    float s2 = 0.f;
    for (int s = 0; s < 128; ++s) s2 += bf2f(Ks[s][t]);
    atomicAdd(ksum + bh * DIMc + t, s2);
  }
}

// ============================================================
// rnorm[b,s,h] = 1 / (sum_m Qf[s, h*DIM+m] * ksum[b,h,m] + eps)
// (reciprocal stored so the GEMM epilogue is a single multiply)
// ============================================================
__global__ void norm_kernel(const unsigned short* __restrict__ Qf,
                            const float* __restrict__ ksum,
                            float* __restrict__ rnorm) {
  int i = blockIdx.x * blockDim.x + threadIdx.x;   // ROWS * H
  if (i >= ROWS * Hc) return;
  int s = i >> 4, h = i & 15;
  int b = s >> 12;
  const unsigned short* q = Qf + (size_t)s * KDq + h * DIMc;
  const float* ks = ksum + (b * Hc + h) * DIMc;
  float acc = 1e-6f;
#pragma unroll 8
  for (int m = 0; m < DIMc; ++m) acc += bf2f(q[m]) * ks[m];
  rnorm[i] = 1.0f / acc;
}

// ============================================================
// O[s, h*DH+d] = (sum_m Qf[s,h*DIM+m] * kvT[b,h,d,m]) * rnorm[s,h] -> bf16
// 2x4 blocked WMMA, K = 64 (2 k-steps); 64-col strip stays inside one head.
// ============================================================
__global__ __launch_bounds__(256) void gemm_attn(
    const unsigned short* __restrict__ Qf,     // [ROWS, KDq]
    const unsigned short* __restrict__ kvTb,   // [B,H,DH,DIM] bf16
    const float* __restrict__ rnorm,           // [ROWS, H]
    unsigned short* __restrict__ Obf) {        // [ROWS, HIDc]
  const int lane = threadIdx.x & 31;
  const int widx = blockIdx.x * 8 + (threadIdx.x >> 5);
  const int tilesN = HIDc / 64;                // 32
  const int tm = widx / tilesN, tn = widx % tilesN;
  const int l15 = lane & 15, hi = (lane >> 4) & 1;
  const int h    = (tn * 64) >> 7;             // head (64-col strip in one head)
  const int b    = (tm * 32) >> 12;            // batch (32-row strip in one batch)
  const int dloc = (tn * 64) & 127;            // d-offset inside head (0 or 64)

  const unsigned short* ap0 =
      Qf + (size_t)(tm * 32 + l15) * KDq + h * DIMc + hi * 8;
  const unsigned short* ap1 = ap0 + (size_t)16 * KDq;
  const unsigned short* bp0 =
      kvTb + ((size_t)(b * Hc + h) * DHc + dloc + l15) * DIMc + hi * 8;

  v8f acc[2][4] = {};
#pragma unroll
  for (int k0 = 0; k0 < DIMc; k0 += 32) {
    v16bf af[2], bfr[4];
    {
      U4x2 u;
      u.lo = *(const uint4*)(ap0 + k0);
      u.hi = *(const uint4*)(ap0 + k0 + 16);
      af[0] = __builtin_bit_cast(v16bf, u);
      u.lo = *(const uint4*)(ap1 + k0);
      u.hi = *(const uint4*)(ap1 + k0 + 16);
      af[1] = __builtin_bit_cast(v16bf, u);
    }
#pragma unroll
    for (int j = 0; j < 4; ++j) {
      const unsigned short* bp = bp0 + (size_t)(j * 16) * DIMc;
      U4x2 u;
      u.lo = *(const uint4*)(bp + k0);
      u.hi = *(const uint4*)(bp + k0 + 16);
      bfr[j] = __builtin_bit_cast(v16bf, u);
    }
#pragma unroll
    for (int i = 0; i < 2; ++i)
#pragma unroll
      for (int j = 0; j < 4; ++j)
        acc[i][j] = __builtin_amdgcn_wmma_f32_16x16x32_bf16(
            false, af[i], false, bfr[j], (short)0, acc[i][j], false, false);
  }

#pragma unroll
  for (int j = 0; j < 4; ++j) {
    const int n = tn * 64 + j * 16 + l15;
#pragma unroll
    for (int i = 0; i < 2; ++i) {
      const int row0 = tm * 32 + i * 16 + hi * 8;
#pragma unroll
      for (int v = 0; v < 8; ++v) {
        int row = row0 + v;
        float y = acc[i][j][v] * rnorm[row * Hc + h];
        Obf[(size_t)row * HIDc + n] = f2bf(y);
      }
    }
  }
}

// ============================================================
// host-side launcher
// ============================================================
extern "C" void kernel_launch(void* const* d_in, const int* in_sizes, int n_in,
                              void* d_out, int out_size, void* d_ws, size_t ws_size,
                              hipStream_t stream) {
  const float* hidden = (const float*)d_in[0];
  const float* wq  = (const float*)d_in[1];
  const float* bq  = (const float*)d_in[2];
  const float* wk  = (const float*)d_in[3];
  const float* bk  = (const float*)d_in[4];
  const float* wv  = (const float*)d_in[5];
  const float* bv  = (const float*)d_in[6];
  const float* wo  = (const float*)d_in[7];
  const float* bo  = (const float*)d_in[8];
  const float* wmq = (const float*)d_in[9];
  const float* wmk = (const float*)d_in[10];
  float* out = (float*)d_out;

  char* ws = (char*)d_ws;
  size_t off = 0;
  auto alloc = [&](size_t bytes) {
    size_t o = off;
    off += (bytes + 255) & ~(size_t)255;
    return o;
  };
  unsigned short* hid_bf  = (unsigned short*)(ws + alloc((size_t)ROWS * HIDc * 2));
  unsigned short* v_bf    = (unsigned short*)(ws + alloc((size_t)ROWS * HIDc * 2));
  unsigned short* attn_bf = (unsigned short*)(ws + alloc((size_t)ROWS * HIDc * 2));
  unsigned short* qf      = (unsigned short*)(ws + alloc((size_t)ROWS * KDq * 2));
  unsigned short* kf      = (unsigned short*)(ws + alloc((size_t)ROWS * KDq * 2));
  unsigned short* wq_eff  = (unsigned short*)(ws + alloc((size_t)KDq * HIDc * 2));
  unsigned short* wk_eff  = (unsigned short*)(ws + alloc((size_t)KDq * HIDc * 2));
  unsigned short* wv_bf   = (unsigned short*)(ws + alloc((size_t)HIDc * HIDc * 2));
  unsigned short* wo_bf   = (unsigned short*)(ws + alloc((size_t)HIDc * HIDc * 2));
  float* kvT_f   = (float*)(ws + alloc((size_t)Bc * Hc * DHc * DIMc * 4));
  float* ksum_f  = (float*)(ws + alloc((size_t)Bc * Hc * DIMc * 4));   // contiguous after kvT_f
  unsigned short* kvT_bf = (unsigned short*)(ws + alloc((size_t)Bc * Hc * DHc * DIMc * 2));
  float* rnorm_f = (float*)(ws + alloc((size_t)ROWS * Hc * 4));
  float* bq_eff  = (float*)(ws + alloc(KDq * 4));
  float* bk_eff  = (float*)(ws + alloc(KDq * 4));

  const int T = 256;
  // 1) casts + weight folding
  cast_f32_bf16<<<(ROWS * HIDc) / T, T, 0, stream>>>(hidden, hid_bf, ROWS * HIDc);
  cast_f32_bf16<<<(HIDc * HIDc) / T, T, 0, stream>>>(wv, wv_bf, HIDc * HIDc);
  cast_f32_bf16<<<(HIDc * HIDc) / T, T, 0, stream>>>(wo, wo_bf, HIDc * HIDc);
  fold_weight<<<(KDq * HIDc) / T, T, 0, stream>>>(wq, wmq, wq_eff);
  fold_weight<<<(KDq * HIDc) / T, T, 0, stream>>>(wk, wmk, wk_eff);
  fold_bias<<<(KDq + T - 1) / T, T, 0, stream>>>(bq, wmq, bq_eff);
  fold_bias<<<(KDq + T - 1) / T, T, 0, stream>>>(bk, wmk, bk_eff);

  // 2) Qf / Kf feature GEMMs (elu+1 fused), V GEMM
  {
    int tilesN = KDq / 64;                        // 16
    int blocks = (ROWS / 32) * tilesN / 8;        // 1024
    gemm_nt<0><<<blocks, T, 0, stream>>>(hid_bf, HIDc, wq_eff, HIDc, bq_eff,
                                         qf, KDq, tilesN, HIDc);
    gemm_nt<0><<<blocks, T, 0, stream>>>(hid_bf, HIDc, wk_eff, HIDc, bk_eff,
                                         kf, KDq, tilesN, HIDc);
  }
  {
    int tilesN = HIDc / 64;                       // 32
    int blocks = (ROWS / 32) * tilesN / 8;        // 2048
    gemm_nt<1><<<blocks, T, 0, stream>>>(hid_bf, HIDc, wv_bf, HIDc, bv,
                                         v_bf, HIDc, tilesN, HIDc);
  }

  // 3) kv^T and ksum (clear accumulators first — graph-replay safe)
  {
    int nz = Bc * Hc * DHc * DIMc + Bc * Hc * DIMc;   // kvT_f + ksum_f contiguous
    zero_f32<<<(nz + T - 1) / T, T, 0, stream>>>(kvT_f, nz);
    kv_ksum<<<Bc * Hc * 32, T, 0, stream>>>(kf, v_bf, kvT_f, ksum_f);
    cast_f32_bf16<<<(Bc * Hc * DHc * DIMc) / T, T, 0, stream>>>(
        kvT_f, kvT_bf, Bc * Hc * DHc * DIMc);
  }

  // 4) norms (reciprocal), attention output GEMM, final projection
  norm_kernel<<<(ROWS * Hc) / T, T, 0, stream>>>(qf, ksum_f, rnorm_f);
  {
    int blocks = (ROWS / 32) * (HIDc / 64) / 8;   // 2048
    gemm_attn<<<blocks, T, 0, stream>>>(qf, kvT_bf, rnorm_f, attn_bf);
    gemm_nt<2><<<blocks, T, 0, stream>>>(attn_bf, HIDc, wo_bf, HIDc, bo,
                                         out, HIDc, HIDc / 64, HIDc);
  }
}
